// StructureGNNEncoder_77120432767650
// MI455X (gfx1250) — compile-verified, matrix operand order checked by reading, not run on hardware
//
#include <hip/hip_runtime.h>
#include <hip/hip_bf16.h>

// ---------------------------------------------------------------------------
// Types for CDNA5 WMMA (wave32, 16x16x32 bf16 -> f32)
// ---------------------------------------------------------------------------
typedef __attribute__((ext_vector_type(16))) __bf16 v16bf;
typedef __attribute__((ext_vector_type(8)))  float  v8f;

union Frag16 { v16bf v; unsigned u[8]; };

__device__ __forceinline__ unsigned short f2bf(float f) {
    unsigned b = __float_as_uint(f);
    unsigned r = (b + 0x7FFFu + ((b >> 16) & 1u)) >> 16;   // RNE
    return (unsigned short)r;
}

// order-preserving float<->uint key for atomic segment-max
__device__ __forceinline__ unsigned fkey(float f) {
    unsigned b = __float_as_uint(f);
    return (b & 0x80000000u) ? ~b : (b | 0x80000000u);
}
__device__ __forceinline__ float funkey(unsigned u) {
    unsigned b = (u & 0x80000000u) ? (u ^ 0x80000000u) : ~u;
    return __uint_as_float(b);
}

__device__ __forceinline__ float wave_sum32(float p) {
    #pragma unroll
    for (int off = 16; off; off >>= 1) p += __shfl_xor(p, off, 32);
    return p;
}

// ---------------------------------------------------------------------------
// Weight packing: f32 [K x Ncols] row-major -> WMMA B-fragment bf16 layout.
// Linear dword index: (((kt*nct + ct)*32 + lane)*8 + v)
//   lane 0..31: column n = ct*16 + (lane&15), group g = lane>>4
//   dword v packs K = kt*32 + (v/4)*16 + g*8 + (v%4)*2 + {0,1}
// ---------------------------------------------------------------------------
__global__ void pack_w_kernel(const float* __restrict__ W, unsigned* __restrict__ Wp,
                              int K, int Ncols) {
    int id = blockIdx.x * blockDim.x + threadIdx.x;
    int nct = Ncols >> 4;
    int total = (K >> 5) * nct * 256;
    if (id >= total) return;
    int v    = id & 7;
    int lane = (id >> 3) & 31;
    int ct   = (id >> 8) % nct;
    int kt   = (id >> 8) / nct;
    int g = lane >> 4;
    int n = ct * 16 + (lane & 15);
    int k0 = kt * 32 + ((v >> 2) << 4) + (g << 3) + ((v & 3) << 1);
    unsigned short lo = f2bf(W[(size_t)k0       * Ncols + n]);
    unsigned short hi = f2bf(W[(size_t)(k0 + 1) * Ncols + n]);
    Wp[id] = (unsigned)lo | ((unsigned)hi << 16);
}

// ---------------------------------------------------------------------------
// GEMM: out[M x Ncols] = A_bf16[M x K] * Wpacked + bias, optional GELU,
// optional bf16 copy of the output. One workgroup = 8 waves = 128 rows x 128
// cols; each wave: 16 rows x 8 col-tiles (v8f acc each). A tile staged in LDS
// via CDNA5 async global->LDS copies (ASYNCcnt), fragments read with ds_load.
// ---------------------------------------------------------------------------
__global__ __launch_bounds__(256) void gemm_bf16_wmma(
    const unsigned short* __restrict__ A,   // M x K (bf16 bits)
    const unsigned*       __restrict__ Wp,  // packed weights
    const float*          __restrict__ bias,
    float*                __restrict__ outF, // nullable
    unsigned short*       __restrict__ outB, // nullable (bf16 bits)
    int M, int K, int Ncols, int act)        // act: 0=none, 1=tanh-GELU
{
    __shared__ unsigned short As[128 * 32];  // 8 KB

    const int tid    = threadIdx.x;
    const int waveId = tid >> 5;
    const int lane   = tid & 31;
    const int rowBlock = blockIdx.x * 128;
    const int colBlock = blockIdx.y * 128;
    const int nct = Ncols >> 4;
    const int ct0 = colBlock >> 4;

    v8f acc[8] = {};

    const int nkt = K >> 5;
    // cooperative A-tile staging coords (fixed per thread): 32 bytes/thread
    const int ldRow = tid >> 1;              // 0..127
    const int ldOff = (tid & 1) * 16;        // half-offset within the 32-K tile
    int gRow = rowBlock + ldRow;
    if (gRow >= M) gRow = M - 1;
    const unsigned short* aRow = A + (size_t)gRow * K + ldOff;
    // low 32 bits of a generic LDS pointer == LDS byte offset (aperture form)
    const unsigned ldsDst = (unsigned)(size_t)(void*)&As[ldRow * 32 + ldOff];

    for (int kt = 0; kt < nkt; ++kt) {
        // stage A tile: 128 rows x 32 halves via async global->LDS (no VGPRs)
        {
            unsigned long long gaddr = (unsigned long long)(size_t)(aRow + kt * 32);
            asm volatile(
                "global_load_async_to_lds_b128 %0, %1, off\n\t"
                "global_load_async_to_lds_b128 %0, %1, off offset:16"
                :: "v"(ldsDst), "v"(gaddr) : "memory");
            if (kt + 1 < nkt)
                __builtin_prefetch(aRow + (kt + 1) * 32, 0, 1);  // global_prefetch_b8
            asm volatile("s_wait_asynccnt 0x0" ::: "memory");
        }
        __syncthreads();

        // A fragment for this wave's 16 rows (from LDS, ds_load)
        Frag16 a;
        {
            const int r = (waveId << 4) + (lane & 15);
            const int g = lane >> 4;
            #pragma unroll
            for (int v = 0; v < 8; ++v) {
                int k = ((v >> 2) << 4) + (g << 3) + ((v & 3) << 1);
                a.u[v] = *(const unsigned*)&As[r * 32 + k];
            }
        }

        // 8 column tiles: load B fragment (coalesced 32B/lane) + WMMA
        #pragma unroll
        for (int ct = 0; ct < 8; ++ct) {
            Frag16 b;
            const unsigned* bp = Wp + ((size_t)(kt * nct + (ct0 + ct)) * 32 + lane) * 8;
            #pragma unroll
            for (int v = 0; v < 8; ++v) b.u[v] = bp[v];
            acc[ct] = __builtin_amdgcn_wmma_f32_16x16x32_bf16(
                false, a.v, false, b.v, (short)0, acc[ct], false, false);
        }
        __syncthreads();
    }

    // Epilogue. C/D layout: lane l, vgpr r -> row = r + 8*(l>>4), col = l&15.
    const int rBase = rowBlock + (waveId << 4) + ((lane >> 4) << 3);
    const int cLane = lane & 15;
    #pragma unroll
    for (int ct = 0; ct < 8; ++ct) {
        const int col = colBlock + ct * 16 + cLane;
        const float bv = bias ? bias[col] : 0.0f;
        #pragma unroll
        for (int r = 0; r < 8; ++r) {
            const int row = rBase + r;
            if (row >= M) continue;
            float v = acc[ct][r] + bv;
            if (act == 1) {
                float t = 0.7978845608028654f * (v + 0.044715f * v * v * v);
                v = 0.5f * v * (1.0f + tanhf(t));
            }
            if (outF) outF[(size_t)row * Ncols + col] = v;
            if (outB) outB[(size_t)row * Ncols + col] = f2bf(v);
        }
    }
}

// ---------------------------------------------------------------------------
// Misc small kernels
// ---------------------------------------------------------------------------
__global__ void fill_u32_kernel(unsigned* __restrict__ p, unsigned val, size_t n) {
    size_t i = (size_t)blockIdx.x * blockDim.x + threadIdx.x;
    if (i < n) p[i] = val;
}

__global__ void cast_copy_kernel(const float* __restrict__ x,
                                 float* __restrict__ h,
                                 unsigned short* __restrict__ hb, size_t n) {
    size_t i = (size_t)blockIdx.x * blockDim.x + threadIdx.x;
    if (i < n) { float v = x[i]; h[i] = v; hb[i] = f2bf(v); }
}

// ---------------------------------------------------------------------------
// Edge kernels (wave per edge; Etot = E + N self-loops appended in ref order)
// ---------------------------------------------------------------------------
__global__ __launch_bounds__(256) void edge_score_kernel(
    const long long* __restrict__ ei, int E, int Etot,
    const float* __restrict__ xl, const float* __restrict__ xr,
    const float* __restrict__ att,
    float* __restrict__ esc, unsigned* __restrict__ emax)
{
    int wave = (int)(((size_t)blockIdx.x * blockDim.x + threadIdx.x) >> 5);
    int lane = threadIdx.x & 31;
    if (wave >= Etot) return;
    int s, d;
    if (wave < E) { s = (int)ei[wave]; d = (int)ei[(size_t)E + wave]; }
    else          { s = wave - E;      d = s; }
    const float* pl = xl + (size_t)s * 128;
    const float* pr = xr + (size_t)d * 128;
    #pragma unroll
    for (int h = 0; h < 4; ++h) {
        float v = pl[h * 32 + lane] + pr[h * 32 + lane];
        v = v > 0.0f ? v : 0.2f * v;                    // leaky_relu(0.2)
        float p = wave_sum32(v * att[h * 32 + lane]);   // dot with att head
        if (lane == 0) {
            esc[(size_t)wave * 4 + h] = p;
            atomicMax(&emax[(size_t)d * 4 + h], fkey(p));
        }
    }
}

__global__ void edge_exp_kernel(
    const long long* __restrict__ ei, int E, int Etot,
    float* __restrict__ esc, const unsigned* __restrict__ emax,
    float* __restrict__ denom)
{
    int t = (int)((size_t)blockIdx.x * blockDim.x + threadIdx.x);
    if (t >= Etot * 4) return;
    int e = t >> 2, h = t & 3;
    int d = (e < E) ? (int)ei[(size_t)E + e] : (e - E);
    float m = funkey(emax[(size_t)d * 4 + h]);
    float ex = __expf(esc[t] - m);
    esc[t] = ex;
    atomicAdd(&denom[(size_t)d * 4 + h], ex);
}

__global__ __launch_bounds__(256) void edge_agg_kernel(
    const long long* __restrict__ ei, int E, int Etot,
    const float* __restrict__ xl, const float* __restrict__ esc,
    const float* __restrict__ denom, float* __restrict__ agg)
{
    int wave = (int)(((size_t)blockIdx.x * blockDim.x + threadIdx.x) >> 5);
    int lane = threadIdx.x & 31;
    if (wave >= Etot) return;
    int s, d;
    if (wave < E) { s = (int)ei[wave]; d = (int)ei[(size_t)E + wave]; }
    else          { s = wave - E;      d = s; }
    #pragma unroll
    for (int h = 0; h < 4; ++h) {
        float alpha = esc[(size_t)wave * 4 + h] / denom[(size_t)d * 4 + h];
        atomicAdd(&agg[(size_t)d * 128 + h * 32 + lane],
                  xl[(size_t)s * 128 + h * 32 + lane] * alpha);
    }
}

// ---------------------------------------------------------------------------
// Fused ELU + residual + LayerNorm (wave per node row), emits f32 + bf16
// ---------------------------------------------------------------------------
__global__ __launch_bounds__(256) void gat_post_kernel(
    const float* __restrict__ agg, const float* __restrict__ bias,
    const float* __restrict__ gamma, const float* __restrict__ beta,
    float* __restrict__ h, unsigned short* __restrict__ hb, int Nn)
{
    int wave = (int)(((size_t)blockIdx.x * blockDim.x + threadIdx.x) >> 5);
    int lane = threadIdx.x & 31;
    if (wave >= Nn) return;
    float g[4]; float s = 0.0f;
    #pragma unroll
    for (int j = 0; j < 4; ++j) {
        int f = j * 32 + lane;
        float a = agg[(size_t)wave * 128 + f] + bias[f];
        a = a > 0.0f ? a : (__expf(a) - 1.0f);          // ELU
        g[j] = a + h[(size_t)wave * 128 + f];           // residual
        s += g[j];
    }
    float mean = wave_sum32(s) * (1.0f / 128.0f);
    float vs = 0.0f;
    #pragma unroll
    for (int j = 0; j < 4; ++j) { float dlt = g[j] - mean; vs += dlt * dlt; }
    float rstd = rsqrtf(wave_sum32(vs) * (1.0f / 128.0f) + 1e-5f);
    #pragma unroll
    for (int j = 0; j < 4; ++j) {
        int f = j * 32 + lane;
        float o = gamma[f] * (g[j] - mean) * rstd + beta[f];
        h [(size_t)wave * 128 + f] = o;
        hb[(size_t)wave * 128 + f] = f2bf(o);
    }
}

__global__ __launch_bounds__(256) void final_ln_kernel(
    const float* __restrict__ h, const float* __restrict__ y,
    const float* __restrict__ gamma, const float* __restrict__ beta,
    float* __restrict__ out, int Nn)
{
    int wave = (int)(((size_t)blockIdx.x * blockDim.x + threadIdx.x) >> 5);
    int lane = threadIdx.x & 31;
    if (wave >= Nn) return;
    float g[4]; float s = 0.0f;
    #pragma unroll
    for (int j = 0; j < 4; ++j) {
        int f = j * 32 + lane;
        g[j] = h[(size_t)wave * 128 + f] + y[(size_t)wave * 128 + f];
        s += g[j];
    }
    float mean = wave_sum32(s) * (1.0f / 128.0f);
    float vs = 0.0f;
    #pragma unroll
    for (int j = 0; j < 4; ++j) { float dlt = g[j] - mean; vs += dlt * dlt; }
    float rstd = rsqrtf(wave_sum32(vs) * (1.0f / 128.0f) + 1e-5f);
    #pragma unroll
    for (int j = 0; j < 4; ++j) {
        int f = j * 32 + lane;
        out[(size_t)wave * 128 + f] = gamma[f] * (g[j] - mean) * rstd + beta[f];
    }
}

// ---------------------------------------------------------------------------
// Host-side orchestration
// ---------------------------------------------------------------------------
extern "C" void kernel_launch(void* const* d_in, const int* in_sizes, int n_in,
                              void* d_out, int out_size, void* d_ws, size_t ws_size,
                              hipStream_t stream) {
    (void)n_in; (void)out_size; (void)ws_size;
    const int D = 128, DF = 512;
    const int N    = in_sizes[0] / D;
    const int E    = in_sizes[1] / 2;
    const int Etot = E + N;

    const float*     x  = (const float*)d_in[0];
    const long long* ei = (const long long*)d_in[1];

    struct LayerP { const float *Wl,*bl,*Wr,*br,*att,*bias,*gamma,*beta; } LP[3];
    int ix = 2;
    for (int l = 0; l < 3; ++l) {
        LP[l].Wl   = (const float*)d_in[ix++]; LP[l].bl   = (const float*)d_in[ix++];
        LP[l].Wr   = (const float*)d_in[ix++]; LP[l].br   = (const float*)d_in[ix++];
        LP[l].att  = (const float*)d_in[ix++]; LP[l].bias = (const float*)d_in[ix++];
        LP[l].gamma= (const float*)d_in[ix++]; LP[l].beta = (const float*)d_in[ix++];
    }
    const float* W1 = (const float*)d_in[ix++]; const float* b1 = (const float*)d_in[ix++];
    const float* W2 = (const float*)d_in[ix++]; const float* b2 = (const float*)d_in[ix++];
    const float* gF = (const float*)d_in[ix++]; const float* bF = (const float*)d_in[ix++];

    // workspace carve-up
    char* wp = (char*)d_ws;
    auto carve = [&](size_t bytes) -> void* {
        void* p = (void*)wp; wp += (bytes + 255) & ~(size_t)255; return p;
    };
    float*          h    = (float*)         carve((size_t)N * D * 4);
    unsigned short* hb   = (unsigned short*)carve((size_t)N * D * 2);
    float*          xl   = (float*)         carve((size_t)N * D * 4);
    float*          xr   = (float*)         carve((size_t)N * D * 4);   // reused as y2
    float*          agg  = (float*)         carve((size_t)N * D * 4);
    float*          esc  = (float*)         carve((size_t)Etot * 4 * 4);
    unsigned*       emax = (unsigned*)      carve((size_t)N * 4 * 4);
    float*          denom= (float*)         carve((size_t)N * 4 * 4);
    unsigned short* y1b  = (unsigned short*)carve((size_t)N * DF * 2);
    unsigned*       WlP[3]; unsigned* WrP[3];
    for (int l = 0; l < 3; ++l) {
        WlP[l] = (unsigned*)carve((size_t)D * D * 2);
        WrP[l] = (unsigned*)carve((size_t)D * D * 2);
    }
    unsigned* W1P = (unsigned*)carve((size_t)D * DF * 2);
    unsigned* W2P = (unsigned*)carve((size_t)DF * D * 2);

    // 1) pack weights into WMMA B-fragment layout
    {
        int tot = (D >> 5) * (D >> 4) * 256;        // 128x128
        for (int l = 0; l < 3; ++l) {
            pack_w_kernel<<<(tot + 255) / 256, 256, 0, stream>>>(LP[l].Wl, WlP[l], D, D);
            pack_w_kernel<<<(tot + 255) / 256, 256, 0, stream>>>(LP[l].Wr, WrP[l], D, D);
        }
        int tot1 = (D  >> 5) * (DF >> 4) * 256;
        int tot2 = (DF >> 5) * (D  >> 4) * 256;
        pack_w_kernel<<<(tot1 + 255) / 256, 256, 0, stream>>>(W1, W1P, D, DF);
        pack_w_kernel<<<(tot2 + 255) / 256, 256, 0, stream>>>(W2, W2P, DF, D);
    }

    // 2) h = x (f32) and bf16 copy
    {
        size_t n = (size_t)N * D;
        cast_copy_kernel<<<(unsigned)((n + 255) / 256), 256, 0, stream>>>(x, h, hb, n);
    }

    const int mb = (N + 127) / 128;
    const int eWaveBlocks = (Etot + 7) / 8;
    const int nWaveBlocks = (N + 7) / 8;

    // 3) GAT layers
    for (int l = 0; l < 3; ++l) {
        gemm_bf16_wmma<<<dim3(mb, 1), 256, 0, stream>>>(
            hb, WlP[l], LP[l].bl, xl, nullptr, N, D, D, 0);
        gemm_bf16_wmma<<<dim3(mb, 1), 256, 0, stream>>>(
            hb, WrP[l], LP[l].br, xr, nullptr, N, D, D, 0);

        size_t nh = (size_t)N * 4;
        fill_u32_kernel<<<(unsigned)((nh + 255) / 256), 256, 0, stream>>>(emax, 0u, nh);
        fill_u32_kernel<<<(unsigned)((nh + 255) / 256), 256, 0, stream>>>((unsigned*)denom, 0u, nh);
        size_t na = (size_t)N * D;
        fill_u32_kernel<<<(unsigned)((na + 255) / 256), 256, 0, stream>>>((unsigned*)agg, 0u, na);

        edge_score_kernel<<<eWaveBlocks, 256, 0, stream>>>(ei, E, Etot, xl, xr, LP[l].att, esc, emax);
        edge_exp_kernel<<<(Etot * 4 + 255) / 256, 256, 0, stream>>>(ei, E, Etot, esc, emax, denom);
        edge_agg_kernel<<<eWaveBlocks, 256, 0, stream>>>(ei, E, Etot, xl, esc, denom, agg);

        gat_post_kernel<<<nWaveBlocks, 256, 0, stream>>>(
            agg, LP[l].bias, LP[l].gamma, LP[l].beta, h, hb, N);
    }

    // 4) FFN: y1 = gelu(h @ W1 + b1) (bf16 only), y2 = y1 @ W2 + b2 (into xr)
    gemm_bf16_wmma<<<dim3(mb, DF / 128), 256, 0, stream>>>(
        hb, W1P, b1, nullptr, y1b, N, D, DF, 1);
    gemm_bf16_wmma<<<dim3(mb, 1), 256, 0, stream>>>(
        y1b, W2P, b2, xr, nullptr, N, DF, D, 0);

    // 5) out = layer_norm(h + y2)
    final_ln_kernel<<<nWaveBlocks, 256, 0, stream>>>(h, xr, gF, bF, (float*)d_out, N);
}
